// CTCBeamDecoder_62440234549293
// MI455X (gfx1250) — 8-bit matrix operand order repaired and checked by reading, compiled, not run on hardware
//
#include <hip/hip_runtime.h>
#include <hip/hip_bf16.h>
#include <math.h>

#define BM 16
#define CC 16
#define KPOOL 272
#define KPAD 288
#define VOCAB 32000
#define LOG_ZERO_F (-1e7f)
#define NEG_INF_F  (-1e30f)
#define M1U 1000003u
#define M2U 8191u

typedef int v8i __attribute__((ext_vector_type(8)));

__device__ __forceinline__ float lae(float a, float b) {
  float m = fmaxf(a, b);
  float d = fminf(a, b) - m;
  return m + log1pf(__expf(d));
}
__device__ __forceinline__ unsigned long long fkey(float v, int idx) {
  unsigned u = __float_as_uint(v);
  u = (u & 0x80000000u) ? ~u : (u | 0x80000000u);
  return ((unsigned long long)u << 32) | (unsigned)(0x7FFFFFFF - idx);
}
__device__ __forceinline__ float keyval(unsigned long long k) {
  unsigned u = (unsigned)(k >> 32);
  unsigned b = (u & 0x80000000u) ? (u & 0x7FFFFFFFu) : ~u;
  return __uint_as_float(b);
}
__device__ __forceinline__ int keyidx(unsigned long long k) {
  return 0x7FFFFFFF - (int)(unsigned)(k & 0xFFFFFFFFu);
}
__device__ __forceinline__ int sumsq_bytes(unsigned x) {
  int b0 = x & 0xff, b1 = (x >> 8) & 0xff, b2 = (x >> 16) & 0xff, b3 = (x >> 24) & 0xff;
  return b0 * b0 + b1 * b1 + b2 * b2 + b3 * b3;
}

__global__ __launch_bounds__(256) void ctc_frame_prep(
    const float* __restrict__ logits, float* __restrict__ wLogZ,
    float* __restrict__ wBlank, int* __restrict__ wSkip,
    float* __restrict__ wCval, int* __restrict__ wCtok, int T) {
  __shared__ unsigned long long red[256];
  __shared__ float fred[256];
  __shared__ unsigned long long topk[256 * CC];
  __shared__ float s_mx, s_logZ;

  const int tid = threadIdx.x;
  const size_t t = blockIdx.x;
  const float* row = logits + t * (size_t)VOCAB;

  unsigned long long k = 0ull;
  for (int v = tid; v < VOCAB; v += 256) {
    __builtin_prefetch(row + v + 2048, 0, 0);
    unsigned long long kk = fkey(row[v], v);
    k = kk > k ? kk : k;
  }
  red[tid] = k; __syncthreads();
  for (int s = 128; s > 0; s >>= 1) {
    if (tid < s) { unsigned long long o = red[tid + s]; if (o > red[tid]) red[tid] = o; }
    __syncthreads();
  }
  if (tid == 0) { s_mx = keyval(red[0]); wSkip[t] = (keyidx(red[0]) == 0) ? 1 : 0; }
  __syncthreads();
  const float mx = s_mx;

  float s = 0.f;
  for (int v = tid; v < VOCAB; v += 256) s += __expf(row[v] - mx);
  fred[tid] = s; __syncthreads();
  for (int q = 128; q > 0; q >>= 1) { if (tid < q) fred[tid] += fred[tid + q]; __syncthreads(); }
  if (tid == 0) {
    float lz = mx + __logf(fred[0]);
    s_logZ = lz; wLogZ[t] = lz; wBlank[t] = row[0] - lz;
  }
  __syncthreads();
  const float logZ = s_logZ;

  unsigned long long loc[CC];
#pragma unroll
  for (int q = 0; q < CC; q++) loc[q] = 0ull;
  for (int v = 1 + tid; v < VOCAB; v += 256) {
    unsigned long long kk = fkey(row[v], v);
    if (kk > loc[CC - 1]) {
      int p = CC - 1;
      while (p > 0 && loc[p - 1] < kk) { loc[p] = loc[p - 1]; p--; }
      loc[p] = kk;
    }
  }
#pragma unroll
  for (int q = 0; q < CC; q++) topk[tid * CC + q] = loc[q];
  __syncthreads();

  for (int r = 0; r < CC; r++) {
    unsigned long long m = 0ull;
    for (int q = 0; q < CC; q++) { unsigned long long kk = topk[tid * CC + q]; m = kk > m ? kk : m; }
    red[tid] = m; __syncthreads();
    for (int q = 128; q > 0; q >>= 1) {
      if (tid < q) { unsigned long long o = red[tid + q]; if (o > red[tid]) red[tid] = o; }
      __syncthreads();
    }
    unsigned long long w = red[0];
    for (int q = 0; q < CC; q++) if (topk[tid * CC + q] == w) topk[tid * CC + q] = 0ull;
    if (tid == 0) { wCval[t * CC + r] = keyval(w) - logZ; wCtok[t * CC + r] = keyidx(w); }
    __syncthreads();
  }
}

__global__ __launch_bounds__(256) void ctc_beam_step(
    const float* __restrict__ logits,
    const float* __restrict__ wLogZ, const float* __restrict__ wBlank,
    const int* __restrict__ wSkip, const float* __restrict__ wCval,
    const int* __restrict__ wCtok, int* __restrict__ wParent,
    int* __restrict__ wApp, float* __restrict__ out, int T) {
  __shared__ float s_pb[BM], s_pnb[BM], s_pbk[BM], s_pnbk[BM];
  __shared__ unsigned s_h1[BM], s_h2[BM], s_ph1[BM], s_ph2[BM];
  __shared__ int s_len[BM], s_last[BM], s_last2[BM], s_valid[BM];
  __shared__ float P_pb[KPAD], P_pnb[KPAD], P_pbk[KPAD], P_pnbk[KPAD], P_score[KPAD];
  __shared__ unsigned P_h1[KPAD], P_h2[KPAD], P_ph1[KPAD], P_ph2[KPAD], P_lp[KPAD];
  __shared__ int P_len[KPAD], P_last[KPAD], P_last2[KPAD], P_vld[KPAD];
  __shared__ int P_par[KPAD], P_app[KPAD], P_norm[KPAD], P_sup[KPAD], P_tak[KPAD];
  __shared__ float F_blank, F_logZ, F_cval[CC];
  __shared__ int F_skip, F_ctok[CC];
  __shared__ int S_start, S_doskip, S_sel[BM];

  const int tid = threadIdx.x;
  const int wave = tid >> 5, lane = tid & 31, hi = lane >> 4, r16 = lane & 15;

#if __has_builtin(__builtin_amdgcn_s_wait_tensorcnt)
  __builtin_amdgcn_s_wait_tensorcnt(0);
#endif

  if (tid < BM) {
    s_len[tid] = 0; s_last[tid] = -1; s_last2[tid] = -1;
    s_h1[tid] = 0u; s_h2[tid] = 0u; s_ph1[tid] = 0u; s_ph2[tid] = 0u;
    s_pb[tid]  = (tid == 0) ? 0.f : LOG_ZERO_F; s_pnb[tid]  = LOG_ZERO_F;
    s_pbk[tid] = (tid == 0) ? 0.f : LOG_ZERO_F; s_pnbk[tid] = LOG_ZERO_F;
    s_valid[tid] = (tid == 0) ? 1 : 0;
  }
  if (tid == 0) S_start = 1;
  __syncthreads();

  for (int t = 0; t < T; t++) {
    if (tid < CC) { F_cval[tid] = wCval[t * CC + tid]; F_ctok[tid] = wCtok[t * CC + tid]; }
    if (tid == 0) {
      F_blank = wBlank[t]; F_logZ = wLogZ[t]; F_skip = wSkip[t];
      S_doskip = (S_start && F_skip) ? 1 : 0;
      S_start = S_doskip;
    }
    __syncthreads();
    if (S_doskip) {
      if (tid < BM) { wParent[t * BM + tid] = tid; wApp[t * BM + tid] = -1; }
      __syncthreads();
      continue;
    }
    const int is_last = (t == T - 1);

    if (tid < BM) {
      int i = tid;
      int len = s_len[i], last = s_last[i], last2 = s_last2[i], vld = s_valid[i];
      int hasl = len > 0;
      int eos = vld && hasl && (last == 1);
      float clast = logits[(size_t)t * VOCAB + (last > 0 ? last : 0)] - F_logZ;
      float nb1 = hasl ? (s_pnb[i] + clast) : s_pnb[i];
      int jm = -1;
      if (vld && hasl) {
        for (int j = 0; j < BM; j++) {
          if (j == i) continue;
          if (s_valid[j] && s_h1[j] == s_ph1[i] && s_h2[j] == s_ph2[i] && s_len[j] == len - 1) { jm = j; break; }
        }
      }
      float nb2 = nb1;
      if (jm >= 0) {
        int rep = (len >= 2) && (last == last2);
        float pref = clast + (rep ? s_pb[jm] : lae(s_pb[jm], s_pnb[jm]));
        nb2 = lae(nb1, pref);
      }
      float bnew = lae(s_pnbk[i], s_pbk[i]) + F_blank;
      float cpnb = eos ? s_pnb[i] : nb2;
      float cpb  = eos ? s_pb[i]  : bnew;
      float cpbk = eos ? s_pbk[i] : cpb;
      float cpnbk= eos ? s_pnbk[i]: cpnb;
      P_pb[i] = cpb; P_pnb[i] = cpnb; P_pbk[i] = cpbk; P_pnbk[i] = cpnbk;
      P_len[i] = len; P_last[i] = last; P_last2[i] = last2;
      P_h1[i] = s_h1[i]; P_h2[i] = s_h2[i]; P_ph1[i] = s_ph1[i]; P_ph2[i] = s_ph2[i];
      P_vld[i] = vld; P_par[i] = i; P_app[i] = -1;
      int pp = KPOOL + i;
      P_pb[pp] = LOG_ZERO_F; P_pnb[pp] = LOG_ZERO_F; P_pbk[pp] = LOG_ZERO_F; P_pnbk[pp] = LOG_ZERO_F;
      P_h1[pp] = 0u; P_h2[pp] = 0u; P_ph1[pp] = 0u; P_ph2[pp] = 0u;
      P_len[pp] = 0; P_last[pp] = -1; P_last2[pp] = -1;
      P_vld[pp] = 0; P_par[pp] = 0; P_app[pp] = -1;
    }
    {
      int e = tid, i = e >> 4, c = e & 15;
      int len = s_len[i], last = s_last[i], vld = s_valid[i];
      int tok = F_ctok[c]; float cv = F_cval[c];
      int same = (len > 0) && (last == tok);
      float base = lae(s_pbk[i], s_pnbk[i]);
      float epnb = cv + (same ? s_pbk[i] : base);
      int eos = vld && (len > 0) && (last == 1);
      int p = BM + e;
      P_pb[p] = LOG_ZERO_F; P_pnb[p] = epnb; P_pbk[p] = LOG_ZERO_F; P_pnbk[p] = epnb;
      P_len[p] = len + 1; P_last[p] = tok; P_last2[p] = last;
      P_h1[p] = s_h1[i] * M1U + (unsigned)tok;
      P_h2[p] = s_h2[i] * M2U + (unsigned)tok;
      P_ph1[p] = s_h1[i]; P_ph2[p] = s_h2[i];
      P_vld[p] = (vld && !eos) ? 1 : 0;
      P_par[p] = i; P_app[p] = tok;
    }
    __syncthreads();

    for (int p = tid; p < KPAD; p += 256) {
      P_score[p] = lae(P_pb[p], P_pnb[p]);
      unsigned lp = (unsigned)(P_len[p] & 0xFFFF);
      P_lp[p] = lp;
      P_norm[p] = sumsq_bytes(P_h1[p]) + sumsq_bytes(P_h2[p]) + sumsq_bytes(lp);
      P_sup[p] = 0; P_tak[p] = 0;
    }
    __syncthreads();

    for (int tp = wave; tp < 9 * 18; tp += 8) {
      int ti = tp / 9;
      int tj0 = (tp % 9) * 2, tj1 = tj0 + 1;
      int ib = ti * 16, jb0 = tj0 * 16, jb1 = tj1 * 16;
      int i = ib + r16;
      int   ni = P_norm[i];
      int   vi = P_vld[i];
      float si = P_score[i];

      v8i B = {0, 0, 0, 0, 0, 0, 0, 0};
      B[0] = hi ? (int)P_lp[i] : (int)P_h1[i]; if (!hi) { B[1] = (int)P_h2[i]; }

      v8i A0 = {0, 0, 0, 0, 0, 0, 0, 0};
      v8i A1 = {0, 0, 0, 0, 0, 0, 0, 0};
      A0[0] = hi ? (int)P_lp[jb0 + r16] : (int)P_h1[jb0 + r16];
      A0[1] = hi ? 0 : (int)P_h2[jb0 + r16];
      A1[0] = hi ? (int)P_lp[jb1 + r16] : (int)P_h1[jb1 + r16];
      A1[1] = hi ? 0 : (int)P_h2[jb1 + r16];

      v8i C0 = {0, 0, 0, 0, 0, 0, 0, 0};
      v8i C1 = {0, 0, 0, 0, 0, 0, 0, 0};
      C0 = __builtin_amdgcn_wmma_i32_16x16x64_iu8(false, A0, false, B, C0, false, false);
      C1 = __builtin_amdgcn_wmma_i32_16x16x64_iu8(false, A1, false, B, C1, false, false);

      int sup = 0;
#pragma unroll
      for (int v = 0; v < 8; v++) {
        int j0 = jb0 + v + (hi ? 8 : 0);
        int D0 = ni + P_norm[j0] - 2 * C0[v];
        float sj0 = P_score[j0];
        sup |= (int)(D0 == 0) & (int)(i != j0) & P_vld[j0] &
               ((int)(sj0 > si) | ((int)(sj0 == si) & (int)(j0 < i)));
        int j1 = jb1 + v + (hi ? 8 : 0);
        int D1 = ni + P_norm[j1] - 2 * C1[v];
        float sj1 = P_score[j1];
        sup |= (int)(D1 == 0) & (int)(i != j1) & P_vld[j1] &
               ((int)(sj1 > si) | ((int)(sj1 == si) & (int)(j1 < i)));
      }
      if (sup & vi) P_sup[i] = 1;
    }
    __syncthreads();

    if (tid == 0) {
      for (int r = 0; r < BM; r++) {
        int best = -1; float bs = NEG_INF_F;
        for (int p = 0; p < KPOOL; p++) {
          if (P_tak[p]) continue;
          float sc = (P_vld[p] && !P_sup[p])
                         ? (is_last ? P_score[p] / fmaxf((float)P_len[p], 1.f) : P_score[p])
                         : NEG_INF_F;
          if (best < 0 || sc > bs) { best = p; bs = sc; }
        }
        S_sel[r] = best; P_tak[best] = 1;
      }
    }
    __syncthreads();

    if (tid < BM) {
      int p = S_sel[tid];
      s_pb[tid] = P_pb[p]; s_pnb[tid] = P_pnb[p]; s_pbk[tid] = P_pbk[p]; s_pnbk[tid] = P_pnbk[p];
      s_h1[tid] = P_h1[p]; s_h2[tid] = P_h2[p]; s_ph1[tid] = P_ph1[p]; s_ph2[tid] = P_ph2[p];
      s_len[tid] = P_len[p]; s_last[tid] = P_last[p]; s_last2[tid] = P_last2[p];
      s_valid[tid] = (P_vld[p] && !P_sup[p]) ? 1 : 0;
      wParent[t * BM + tid] = P_par[p]; wApp[t * BM + tid] = P_app[p];
    }
    __syncthreads();
  }

  for (int i = tid; i < BM * T; i += 256) out[i] = -1.f;
  __syncthreads();
  if (tid < BM) {
    int b = tid, len = s_len[b];
    out[(size_t)BM * T + b] = (float)len;
    float sc = s_valid[b] ? lae(s_pb[b], s_pnb[b]) / fmaxf((float)len, 1.f) : LOG_ZERO_F;
    out[(size_t)BM * T + BM + b] = sc;
    int cur = b, pos = len;
    for (int tt = T - 1; tt >= 0; tt--) {
      int a = wApp[tt * BM + cur];
      int par = wParent[tt * BM + cur];
      if (a >= 0) { pos--; if (pos >= 0 && pos < T) out[(size_t)b * T + pos] = (float)a; }
      cur = par;
    }
  }
}

extern "C" void kernel_launch(void* const* d_in, const int* in_sizes, int n_in,
                              void* d_out, int out_size, void* d_ws, size_t ws_size,
                              hipStream_t stream) {
  (void)n_in; (void)out_size; (void)ws_size;
  const float* logits = (const float*)d_in[0];
  int T = in_sizes[0] / VOCAB;

  char* p = (char*)d_ws;
  float* wLogZ  = (float*)p; p += sizeof(float) * T;
  float* wBlank = (float*)p; p += sizeof(float) * T;
  int*   wSkip  = (int*)p;   p += sizeof(int) * T;
  float* wCval  = (float*)p; p += sizeof(float) * T * CC;
  int*   wCtok  = (int*)p;   p += sizeof(int) * T * CC;
  int*   wParent= (int*)p;   p += sizeof(int) * T * BM;
  int*   wApp   = (int*)p;   p += sizeof(int) * T * BM;

  ctc_frame_prep<<<T, 256, 0, stream>>>(logits, wLogZ, wBlank, wSkip, wCval, wCtok, T);
  ctc_beam_step<<<1, 256, 0, stream>>>(logits, wLogZ, wBlank, wSkip, wCval, wCtok,
                                       wParent, wApp, (float*)d_out, T);
}
